// IDCCT2_1133871366834
// MI455X (gfx1250) — matrix operand order repaired and checked by reading, not compiled, not run on hardware
//
#include <hip/hip_runtime.h>

typedef _Float16 v4h  __attribute__((ext_vector_type(4)));
typedef _Float16 v8h  __attribute__((ext_vector_type(8)));
typedef _Float16 v16h __attribute__((ext_vector_type(16)));
typedef float    v8f  __attribute__((ext_vector_type(8)));

#define NDIM 4096
#define BM 128
#define BN 128
#define BK 32
#define LDH 40  // LDS tile row stride in halves (BK + 8 pad): 80B rows, conflict-free b128 frag reads

// cos(pi * a * (2b+1) / 8192); integer phase reduced exactly mod 16384.
__device__ __forceinline__ float cos_tw(unsigned a, unsigned b) {
    unsigned m = (a * (2u * b + 1u)) & 16383u;
    return __cosf((float)m * 3.83495196971410329e-4f);  // pi/8192
}

// f32 -> f16 hi/lo split: f ~= (float)h + (float)l, ~22 mantissa bits combined.
__device__ __forceinline__ void split2(float f, _Float16& h, _Float16& l) {
    h = (_Float16)f;
    l = (_Float16)(f - (float)h);
}

__device__ __forceinline__ v16h cat8(v8h a, v8h b) {
    return __builtin_shufflevector(a, b, 0, 1, 2, 3, 4, 5, 6, 7,
                                   8, 9, 10, 11, 12, 13, 14, 15);
}

// GEN_A = false : D = G * C^T        (pass 1: A = X from global, B generated)
// GEN_A = true  : D = Cgen * G       (pass 2: A generated,       B = Y from global)
template <bool GEN_A>
__global__ __launch_bounds__(256) void dct_gemm(const float* __restrict__ G,
                                                float* __restrict__ D) {
    __shared__ __align__(16) _Float16 Ah[BM * LDH];  // A hi tile, [m][k]
    __shared__ __align__(16) _Float16 Al[BM * LDH];  // A lo tile
    __shared__ __align__(16) _Float16 Bh[BN * LDH];  // B hi tile, transposed [n][k]
    __shared__ __align__(16) _Float16 Bl[BN * LDH];  // B lo tile

    const int tid  = threadIdx.x;
    const int lane = tid & 31;
    const int wave = tid >> 5;
    const int wm   = (wave & 1) * 64;   // wave row offset inside block tile
    const int wn   = (wave >> 1) * 32;  // wave col offset inside block tile
    const int bm   = blockIdx.y * BM;
    const int bn   = blockIdx.x * BN;
    const int l16  = lane & 15;
    const int hi   = lane >> 4;

    const v8f vzero = {};
    v8f acc[4][2];
#pragma unroll
    for (int i = 0; i < 4; ++i)
#pragma unroll
        for (int j = 0; j < 2; ++j) acc[i][j] = vzero;

    for (int k0 = 0; k0 < NDIM; k0 += BK) {
        if (GEN_A) {
            // A tile generated: A[m][k] = cos(pi*(k0+k)*(2*(bm+m)+1)/8192)
            const int m = tid >> 1, kh = (tid & 1) * 16;
#pragma unroll
            for (int q = 0; q < 4; ++q) {
                v4h hh, ll;
#pragma unroll
                for (int e = 0; e < 4; ++e) {
                    float f = cos_tw((unsigned)(k0 + kh + q * 4 + e),
                                     (unsigned)(bm + m));
                    _Float16 h, l;
                    split2(f, h, l);
                    hh[e] = h;
                    ll[e] = l;
                }
                *(v4h*)&Ah[m * LDH + kh + q * 4] = hh;
                *(v4h*)&Al[m * LDH + kh + q * 4] = ll;
            }
            // B tile from global Y: rows p = k0..k0+31, cols v = bn..bn+127
            const int r = tid >> 3, c = (tid & 7) * 16;
            const float4* src =
                (const float4*)(G + (size_t)(k0 + r) * NDIM + bn + c);
#pragma unroll
            for (int q = 0; q < 4; ++q) {
                float4 v = src[q];
                float f[4] = {v.x, v.y, v.z, v.w};
#pragma unroll
                for (int e = 0; e < 4; ++e) {
                    _Float16 h, l;
                    split2(f[e], h, l);
                    Bh[(c + q * 4 + e) * LDH + r] = h;
                    Bl[(c + q * 4 + e) * LDH + r] = l;
                }
            }
            if (k0 + BK < NDIM)
                __builtin_prefetch(G + (size_t)(k0 + BK + r) * NDIM + bn + c, 0, 0);
        } else {
            // A tile from global X: rows m = bm.., cols k = k0..
            const int m = tid >> 1, kh = (tid & 1) * 16;
            const float4* src =
                (const float4*)(G + (size_t)(bm + m) * NDIM + k0 + kh);
#pragma unroll
            for (int q = 0; q < 4; ++q) {
                float4 v = src[q];
                float f[4] = {v.x, v.y, v.z, v.w};
                v4h hh, ll;
#pragma unroll
                for (int e = 0; e < 4; ++e) {
                    _Float16 h, l;
                    split2(f[e], h, l);
                    hh[e] = h;
                    ll[e] = l;
                }
                *(v4h*)&Ah[m * LDH + kh + q * 4] = hh;
                *(v4h*)&Al[m * LDH + kh + q * 4] = ll;
            }
            // B tile generated: B[n][k] = cos(pi*(k0+k)*(2*(bn+n)+1)/8192)
#pragma unroll
            for (int q = 0; q < 4; ++q) {
                v4h hh, ll;
#pragma unroll
                for (int e = 0; e < 4; ++e) {
                    float f = cos_tw((unsigned)(k0 + kh + q * 4 + e),
                                     (unsigned)(bn + m));
                    _Float16 h, l;
                    split2(f, h, l);
                    hh[e] = h;
                    ll[e] = l;
                }
                *(v4h*)&Bh[m * LDH + kh + q * 4] = hh;
                *(v4h*)&Bl[m * LDH + kh + q * 4] = ll;
            }
            if (k0 + BK < NDIM)
                __builtin_prefetch(G + (size_t)(bm + m) * NDIM + k0 + BK + kh, 0, 0);
        }
        __syncthreads();

        // One K=32 step: B frags hoisted, A frags streamed; 3 WMMAs per tile
        // (Ah*Bh + Ah*Bl + Al*Bh), lo*lo dropped (~2^-22 relative).
        v16h bfh[2], bfl[2];
#pragma unroll
        for (int j = 0; j < 2; ++j) {
            const _Float16* p = &Bh[(wn + j * 16 + l16) * LDH + 16 * hi];
            bfh[j] = cat8(*(const v8h*)p, *(const v8h*)(p + 8));
            const _Float16* q = &Bl[(wn + j * 16 + l16) * LDH + 16 * hi];
            bfl[j] = cat8(*(const v8h*)q, *(const v8h*)(q + 8));
        }
#pragma unroll
        for (int i = 0; i < 4; ++i) {
            const _Float16* p = &Ah[(wm + i * 16 + l16) * LDH + 16 * hi];
            v16h afh = cat8(*(const v8h*)p, *(const v8h*)(p + 8));
            const _Float16* q = &Al[(wm + i * 16 + l16) * LDH + 16 * hi];
            v16h afl = cat8(*(const v8h*)q, *(const v8h*)(q + 8));
#pragma unroll
            for (int j = 0; j < 2; ++j) {
                acc[i][j] = __builtin_amdgcn_wmma_f32_16x16x32_f16(
                    false, afh, false, bfh[j], (short)0, acc[i][j], false, false);
                acc[i][j] = __builtin_amdgcn_wmma_f32_16x16x32_f16(
                    false, afh, false, bfl[j], (short)0, acc[i][j], false, false);
                acc[i][j] = __builtin_amdgcn_wmma_f32_16x16x32_f16(
                    false, afl, false, bfh[j], (short)0, acc[i][j], false, false);
            }
        }
        __syncthreads();
    }

    // Store: C/D layout = VGPR r -> rows (r | r+8) by half-wave, col = lane%16
#pragma unroll
    for (int i = 0; i < 4; ++i) {
#pragma unroll
        for (int j = 0; j < 2; ++j) {
            const int col = bn + wn + j * 16 + l16;
#pragma unroll
            for (int r = 0; r < 8; ++r) {
                const int row = bm + wm + i * 16 + r + 8 * hi;
                D[(size_t)row * NDIM + col] = acc[i][j][r];
            }
        }
    }
}

extern "C" void kernel_launch(void* const* d_in, const int* in_sizes, int n_in,
                              void* d_out, int out_size, void* d_ws, size_t ws_size,
                              hipStream_t stream) {
    const float* x = (const float*)d_in[0];
    float* z = (float*)d_out;
    float* y = (float*)d_ws;  // 4096*4096*4 = 64 MB intermediate

    dim3 grid(NDIM / BN, NDIM / BM);
    dim3 block(256);
    // Pass 1: Y = X * C^T   (transform along last axis)
    dct_gemm<false><<<grid, block, 0, stream>>>(x, y);
    // Pass 2: Z = C * Y     (transform along first axis)
    dct_gemm<true><<<grid, block, 0, stream>>>(y, z);
}
